// GNN_10479720202600
// MI455X (gfx1250) — matrix operand order repaired and checked by reading, not compiled, not run on hardware
//
#include <hip/hip_runtime.h>
#include <hip/hip_bf16.h>

// ---------------- problem constants ----------------
#define N_NODES 100000
#define N_EDGES 800000
#define NGRP    16
#define EDIM    164
#define TWIN    5
#define NLAY    6
#define NPAD    176        // 164 padded to 11 * 16
#define HPAD    176        // padded bf16 h row length
#define KBIG    384        // [h_dst 176][h_src|agg 176][extras 8 | 1.0 | pad] = 12 * 32
#define KMID    192        // 164 pad | 1.0 at 176 | pad = 6 * 32
#define KENC    32         // 8 real | 1.0 at 8 | pad = 1 * 32
#define NT_MSG  (N_EDGES / 16)   // 50000 edge tiles
#define NT_UPD  (N_NODES / 16)   // 6250 node tiles

// dynamic LDS partition (bf16 elements)
#define SW1_ELEMS  (NPAD * KBIG)          // 67584
#define SW2_ELEMS  (NPAD * KMID)          // 33792
#define SA_OFF     (SW1_ELEMS + SW2_ELEMS)        // 101376
#define SM_OFF     (SA_OFF + 4 * 16 * KBIG)       // 125952
#define SH_ELEMS   (SM_OFF + 4 * 16 * KMID)       // 138240
#define SH_BYTES   (SH_ELEMS * 2)                 // 276480 < 320KB WGP LDS

typedef __bf16 bf16_t;
typedef __attribute__((ext_vector_type(4)))  float  v4f;
typedef __attribute__((ext_vector_type(8)))  __bf16 v8bf;
typedef __attribute__((ext_vector_type(16))) __bf16 v16bf;
typedef __attribute__((ext_vector_type(8)))  float  v8f;

union FragA { v16bf v; v8bf h[2]; };

__device__ __forceinline__ float silu_f(float x) {
  return x * __builtin_amdgcn_rcpf(1.0f + __expf(-x));   // 1 v_rcp instead of div chain
}

// ---------------------------------------------------------------------------
// One 16x16 output tile of A[16xK] * W^T, bias pre-folded into W via a
// constant-1.0 input column. Accumulator starts at 0 -> inline SRC2=0 WMMA.
// A row-major [16][KPAD] bf16 (LDS), W [NPAD][KPAD] bf16 (LDS or global).
// B-fragment: lanes 0-15 -> N=lane, K=kc*32+0..15 ; lanes 16-31 -> K=+16..31.
// A-fragment: lanes 0-15 -> M=lane, K=kc*32+0..15 ; lanes 16-31 -> K=+16..31.
// ---------------------------------------------------------------------------
template<int KC, int KPAD>
__device__ __forceinline__ void mlp_tile(const bf16_t* __restrict__ A,
                                         const bf16_t* __restrict__ W,
                                         int lane, int nt, float out[8]) {
  const int col = nt * 16 + (lane & 15);
  v8f acc = {};
#pragma unroll
  for (int kc = 0; kc < KC; ++kc) {
    const int ka = kc * 32 + ((lane & 16) ? 16 : 0);
    const bf16_t* pa = A + (lane & 15) * KPAD + ka;
    FragA fa;
    fa.h[0] = *(const v8bf*)(pa);
    fa.h[1] = *(const v8bf*)(pa + 8);
    const int kb = kc * 32 + ((lane & 16) ? 16 : 0);
    const v16bf b = *(const v16bf*)(W + col * KPAD + kb);
    acc = __builtin_amdgcn_wmma_f32_16x16x32_bf16(false, fa.v, false, b,
                                                  (short)0, acc, false, false);
  }
#pragma unroll
  for (int i = 0; i < 8; ++i) out[i] = acc[i];
}

// ---------------- small utility kernels ----------------
__global__ void zero_kernel(float* p, int n) {
  for (int i = blockIdx.x * blockDim.x + threadIdx.x; i < n; i += gridDim.x * blockDim.x)
    p[i] = 0.0f;
}

__global__ void posmax_kernel(const float* __restrict__ pos, unsigned int* pm) {
  int i = blockIdx.x * blockDim.x + threadIdx.x;
  if (i < N_NODES) atomicMax(pm, __float_as_uint(pos[i]));   // pos >= 0 (uniform)
}

__global__ void deg_kernel(const int* __restrict__ edst, float* __restrict__ deg) {
  int e = blockIdx.x * blockDim.x + threadIdx.x;
  if (e < N_EDGES) atomicAdd(&deg[edst[e]], 1.0f);
}

__global__ void cnt_kernel(const int* __restrict__ batch, float* __restrict__ cnt) {
  int i = blockIdx.x * blockDim.x + threadIdx.x;
  if (i < N_NODES) atomicAdd(&cnt[batch[i]], 1.0f);
}

// f32 weight (L, K, 164) + bias (L, 164) -> bf16 padded transposed (L, NPAD, Kpad).
// The bias vector is written at K-row bias_k (multiplied by the constant-1 column).
// remap!=0 re-layouts K for the fused inputs:
//   src k<164 -> k ; 164..327 -> k+12 ; >=328 -> k+24  (pads in between are zero)
__global__ void prep_kernel(const float* __restrict__ src, const float* __restrict__ bias,
                            bf16_t* __restrict__ dst,
                            int K, int Kpad, int bias_k, int total, int remap) {
  int i = blockIdx.x * blockDim.x + threadIdx.x;
  if (i >= total) return;
  int k = i % Kpad;
  int n = (i / Kpad) % NPAD;
  int l = i / (Kpad * NPAD);
  float v = 0.0f;
  if (k == bias_k) {
    v = (n < EDIM) ? bias[l * EDIM + n] : 0.0f;
  } else {
    int sk; bool valid;
    if (remap) {
      if (k < 164)      { sk = k;      valid = true; }
      else if (k < 176) { sk = 0;      valid = false; }
      else if (k < 340) { sk = k - 12; valid = true; }
      else if (k < 352) { sk = 0;      valid = false; }
      else              { sk = k - 24; valid = (sk < K); }
    } else { sk = k; valid = (k < K); }
    if (n < EDIM && valid && sk < K) v = src[(long)l * K * EDIM + (long)sk * EDIM + n];
  }
  dst[i] = (bf16_t)v;
}

// ---------------- encoder: h = silu(silu(x W1 + b1) W2 + b2) ----------------
__global__ __launch_bounds__(64) void enc_kernel(
    const float* __restrict__ dataA, const float* __restrict__ posA,
    const float* __restrict__ timeA, const float* __restrict__ varsA,
    const bf16_t* __restrict__ W1, const bf16_t* __restrict__ W2,
    float* __restrict__ h, bf16_t* __restrict__ hbf,
    const unsigned int* __restrict__ pmaxb) {
  __shared__ bf16_t sA[2][16 * KENC];
  __shared__ bf16_t sM[2][16 * KMID];
  const int wave = threadIdx.x >> 5, lane = threadIdx.x & 31;
  const int tile = blockIdx.x * 2 + wave;
  const int row = lane & 15, half = lane >> 4;
  const int node = tile * 16 + row;
  const float pmax = __uint_as_float(*pmaxb);
  bf16_t* A = sA[wave];
  for (int j = 0; j < 16; ++j) {
    int c = half * 16 + j;
    float v = 0.0f;
    if (c < 5)       v = dataA[node * TWIN + c];
    else if (c == 5) v = posA[node] / pmax;
    else if (c == 6) v = timeA[node] * (1.0f / 16.0f);
    else if (c == 7) v = varsA[node];
    else if (c == 8) v = 1.0f;                 // bias column
    A[row * KENC + c] = (bf16_t)v;
  }
  bf16_t* M = sM[wave];
  for (int j = 0; j < 8; ++j) {
    int t = lane * 8 + j;                      // col 176 = 1.0 (bias), rest of pad = 0
    M[(t >> 4) * KMID + 176 + (t & 15)] = (bf16_t)(((t & 15) == 0) ? 1.0f : 0.0f);
  }
  for (int nt = 0; nt < 11; ++nt) {
    float out[8];
    mlp_tile<1, KENC>(A, W1, lane, nt, out);
    int colg = nt * 16 + (lane & 15);
    int rb = (lane & 16) ? 8 : 0;
#pragma unroll
    for (int r = 0; r < 8; ++r) M[(rb + r) * KMID + colg] = (bf16_t)silu_f(out[r]);
  }
  for (int nt = 0; nt < 11; ++nt) {
    float out[8];
    mlp_tile<6, KMID>(M, W2, lane, nt, out);
    int colg = nt * 16 + (lane & 15);
    int rb = (lane & 16) ? 8 : 0;
#pragma unroll
    for (int r = 0; r < 8; ++r) {
      int nn = tile * 16 + rb + r;
      float v = silu_f(out[r]);                 // exactly 0 in pad cols (zero W/bias)
      hbf[(long)nn * HPAD + colg] = (bf16_t)v;
      if (colg < EDIM) h[(long)nn * EDIM + colg] = v;
    }
  }
}

// ---------------- fused message MLP + scatter-add (persistent, weights in LDS) ----
__global__ __launch_bounds__(128) void msg_kernel(
    const bf16_t* __restrict__ hbf, const float* __restrict__ dataA,
    const float* __restrict__ posA, const float* __restrict__ timeA,
    const float* __restrict__ varsA,
    const int* __restrict__ esrc, const int* __restrict__ edst,
    const bf16_t* __restrict__ W1, const bf16_t* __restrict__ W2,
    float* __restrict__ agg, const unsigned int* __restrict__ pmaxb) {
  extern __shared__ bf16_t smem[];
  bf16_t* sW1 = smem;                 // [176][384]
  bf16_t* sW2 = smem + SW1_ELEMS;     // [176][192]
  const int wave = threadIdx.x >> 5, lane = threadIdx.x & 31;
  bf16_t* A = smem + SA_OFF + wave * (16 * KBIG);
  bf16_t* M = smem + SM_OFF + wave * (16 * KMID);
  // stage weights once per workgroup
  for (int i = threadIdx.x; i < SW1_ELEMS / 8; i += 128) ((v8bf*)sW1)[i] = ((const v8bf*)W1)[i];
  for (int i = threadIdx.x; i < SW2_ELEMS / 8; i += 128) ((v8bf*)sW2)[i] = ((const v8bf*)W2)[i];
  // init mid-matrix pad columns once: col 176 = 1.0 (bias), others 0
  for (int j = 0; j < 8; ++j) {
    int t = lane * 8 + j;
    M[(t >> 4) * KMID + 176 + (t & 15)] = (bf16_t)(((t & 15) == 0) ? 1.0f : 0.0f);
  }
  __syncthreads();
  const float pmax = __uint_as_float(*pmaxb);
  const int rb = (lane & 16) ? 8 : 0;
  for (int tile = blockIdx.x * 4 + wave; tile < NT_MSG; tile += gridDim.x * 4) {
    // A = [hbf[dst] 0..175 | hbf[src] 176..351 | extras 352..359 | 1.0 at 360 | 0]
    // 16 rows x 48 vector-chunks of 8 bf16; 768 chunks over 32 lanes = 24 each.
#pragma unroll 4
    for (int i = 0; i < 24; ++i) {
      int ch = lane + 32 * i;
      int r = ch / 48, cc = ch % 48;
      int e = tile * 16 + r;
      v8bf val;
      if (cc < 22) {
        val = *(const v8bf*)(hbf + (long)edst[e] * HPAD + cc * 8);
      } else if (cc < 44) {
        val = *(const v8bf*)(hbf + (long)esrc[e] * HPAD + (cc - 22) * 8);
      } else if (cc == 44) {
        int d = edst[e], s = esrc[e];
#pragma unroll
        for (int j = 0; j < 5; ++j) val[j] = (bf16_t)(dataA[d * TWIN + j] - dataA[s * TWIN + j]);
        val[5] = (bf16_t)((posA[d] - posA[s]) / pmax);
        val[6] = (bf16_t)(timeA[d] * (1.0f / 16.0f));
        val[7] = (bf16_t)(varsA[d]);
      } else if (cc == 45) {                    // bias column at 360
        val[0] = (bf16_t)1.0f;
#pragma unroll
        for (int j = 1; j < 8; ++j) val[j] = (bf16_t)0.0f;
      } else {
#pragma unroll
        for (int j = 0; j < 8; ++j) val[j] = (bf16_t)0.0f;
      }
      *(v8bf*)(A + r * KBIG + cc * 8) = val;
    }
#pragma unroll
    for (int nt = 0; nt < 11; ++nt) {
      float out[8];
      mlp_tile<12, KBIG>(A, sW1, lane, nt, out);
      int colg = nt * 16 + (lane & 15);
#pragma unroll
      for (int r = 0; r < 8; ++r) M[(rb + r) * KMID + colg] = (bf16_t)silu_f(out[r]);
    }
    int dsts[8];
#pragma unroll
    for (int r = 0; r < 8; ++r) dsts[r] = edst[tile * 16 + rb + r];
#pragma unroll
    for (int nt = 0; nt < 11; ++nt) {
      float out[8];
      mlp_tile<6, KMID>(M, sW2, lane, nt, out);
      int colg = nt * 16 + (lane & 15);
      if (colg < EDIM) {
#pragma unroll
        for (int r = 0; r < 8; ++r)
          atomicAdd(&agg[(long)dsts[r] * EDIM + colg], silu_f(out[r]));
      }
    }
  }
}

// ---------------- node update MLP + residual + norm statistics (persistent) ----
__global__ __launch_bounds__(128) void upd_kernel(
    const float* __restrict__ h, const bf16_t* __restrict__ hbf,
    const float* __restrict__ agg, const float* __restrict__ deg,
    const float* __restrict__ timeA, const float* __restrict__ varsA,
    const int* __restrict__ batch,
    const bf16_t* __restrict__ W1, const bf16_t* __restrict__ W2,
    float* __restrict__ hu, float* __restrict__ ssum, float* __restrict__ ssumsq) {
  extern __shared__ bf16_t smem[];
  bf16_t* sW1 = smem;
  bf16_t* sW2 = smem + SW1_ELEMS;
  const int wave = threadIdx.x >> 5, lane = threadIdx.x & 31;
  bf16_t* A = smem + SA_OFF + wave * (16 * KBIG);
  bf16_t* M = smem + SM_OFF + wave * (16 * KMID);
  for (int i = threadIdx.x; i < SW1_ELEMS / 8; i += 128) ((v8bf*)sW1)[i] = ((const v8bf*)W1)[i];
  for (int i = threadIdx.x; i < SW2_ELEMS / 8; i += 128) ((v8bf*)sW2)[i] = ((const v8bf*)W2)[i];
  for (int j = 0; j < 8; ++j) {
    int t = lane * 8 + j;
    M[(t >> 4) * KMID + 176 + (t & 15)] = (bf16_t)(((t & 15) == 0) ? 1.0f : 0.0f);
  }
  __syncthreads();
  const int rb = (lane & 16) ? 8 : 0;
  for (int tile = blockIdx.x * 4 + wave; tile < NT_UPD; tile += gridDim.x * 4) {
    // A = [hbf[node] 0..175 | agg/deg 176..351 | extras 352..353 | 1.0 at 360 | 0]
#pragma unroll 4
    for (int i = 0; i < 24; ++i) {
      int ch = lane + 32 * i;
      int r = ch / 48, cc = ch % 48;
      int node = tile * 16 + r;
      v8bf val;
      if (cc < 22) {
        val = *(const v8bf*)(hbf + (long)node * HPAD + cc * 8);
      } else if (cc < 43) {
        float idg = __builtin_amdgcn_rcpf(fmaxf(deg[node], 1.0f));
        int col0 = (cc - 22) * 8;
        if (cc < 42) {
          v4f a0 = *(const v4f*)(agg + (long)node * EDIM + col0);
          v4f a1 = *(const v4f*)(agg + (long)node * EDIM + col0 + 4);
#pragma unroll
          for (int j = 0; j < 4; ++j) { val[j] = (bf16_t)(a0[j] * idg); val[4 + j] = (bf16_t)(a1[j] * idg); }
        } else {                      // col0 == 160 : cols 160..163 valid, rest pad
          v4f a0 = *(const v4f*)(agg + (long)node * EDIM + col0);
#pragma unroll
          for (int j = 0; j < 4; ++j) { val[j] = (bf16_t)(a0[j] * idg); val[4 + j] = (bf16_t)0.0f; }
        }
      } else if (cc == 44) {
        val[0] = (bf16_t)(timeA[node] * (1.0f / 16.0f));
        val[1] = (bf16_t)(varsA[node]);
#pragma unroll
        for (int j = 2; j < 8; ++j) val[j] = (bf16_t)0.0f;
      } else if (cc == 45) {                    // bias column at 360
        val[0] = (bf16_t)1.0f;
#pragma unroll
        for (int j = 1; j < 8; ++j) val[j] = (bf16_t)0.0f;
      } else {
#pragma unroll
        for (int j = 0; j < 8; ++j) val[j] = (bf16_t)0.0f;
      }
      *(v8bf*)(A + r * KBIG + cc * 8) = val;
    }
#pragma unroll
    for (int nt = 0; nt < 11; ++nt) {
      float out[8];
      mlp_tile<12, KBIG>(A, sW1, lane, nt, out);
      int colg = nt * 16 + (lane & 15);
#pragma unroll
      for (int r = 0; r < 8; ++r) M[(rb + r) * KMID + colg] = (bf16_t)silu_f(out[r]);
    }
    int grp[8];
#pragma unroll
    for (int r = 0; r < 8; ++r) grp[r] = batch[tile * 16 + rb + r];
#pragma unroll
    for (int nt = 0; nt < 11; ++nt) {
      float out[8];
      mlp_tile<6, KMID>(M, sW2, lane, nt, out);
      int colg = nt * 16 + (lane & 15);
      if (colg < EDIM) {
#pragma unroll
        for (int r = 0; r < 8; ++r) {
          int nn = tile * 16 + rb + r;
          float val = h[(long)nn * EDIM + colg] + silu_f(out[r]);
          hu[(long)nn * EDIM + colg] = val;
          atomicAdd(&ssum[grp[r] * EDIM + colg], val);
          atomicAdd(&ssumsq[grp[r] * EDIM + colg], val * val);
        }
      }
    }
  }
}

__global__ void normstat_kernel(const float* __restrict__ ssum, const float* __restrict__ ssumsq,
                                const float* __restrict__ cnt,
                                float* __restrict__ mean, float* __restrict__ rstd) {
  int i = blockIdx.x * blockDim.x + threadIdx.x;
  if (i >= NGRP * EDIM) return;
  int g = i / EDIM;
  float c = fmaxf(cnt[g], 1.0f);
  float m = ssum[i] / c;
  float v = ssumsq[i] / c - m * m;
  mean[i] = m;
  rstd[i] = rsqrtf(v + 1e-5f);
}

__global__ void applynorm_kernel(const float* __restrict__ hu, const int* __restrict__ batch,
                                 const float* __restrict__ mean, const float* __restrict__ rstd,
                                 float* __restrict__ h, bf16_t* __restrict__ hbf) {
  int i = blockIdx.x * blockDim.x + threadIdx.x;
  if (i >= N_NODES * HPAD) return;
  int n = i / HPAD, c = i % HPAD;
  if (c < EDIM) {
    int g = batch[n];
    float v = (hu[(long)n * EDIM + c] - mean[g * EDIM + c]) * rstd[g * EDIM + c];
    h[(long)n * EDIM + c] = v;
    hbf[i] = (bf16_t)v;
  } else {
    hbf[i] = (bf16_t)0.0f;
  }
}

// ---------------- decoder: two strided VALID convs over the channel dim ----------------
__global__ void dec_kernel(const float* __restrict__ h, const float* __restrict__ dataA,
                           const float* __restrict__ dtp,
                           const float* __restrict__ w1, const float* __restrict__ b1,
                           const float* __restrict__ w2, const float* __restrict__ b2,
                           float* __restrict__ out) {
  int n = blockIdx.x * blockDim.x + threadIdx.x;
  if (n >= N_NODES) return;
  const float* hr = h + (long)n * EDIM;
  const float dtv = dtp[0];
  const float last = dataA[n * TWIN + 4];
  for (int t = 0; t < TWIN; ++t) {
    float acc = b2[0];
    for (int ic = 0; ic < 8; ++ic) {
      float bb = b1[ic];
      for (int k = 0; k < 14; ++k) {
        int p1 = t * 9 + k;              // conv1 output position, stride 3, kernel 16
        float a = bb;
#pragma unroll
        for (int k1 = 0; k1 < 16; ++k1) a += hr[p1 * 3 + k1] * w1[ic * 16 + k1];
        acc += silu_f(a) * w2[ic * 14 + k];
      }
    }
    out[n * TWIN + t] = last + dtv * (float)(t + 1) * acc;
  }
}

// ---------------- host orchestration ----------------
extern "C" void kernel_launch(void* const* d_in, const int* in_sizes, int n_in,
                              void* d_out, int out_size, void* d_ws, size_t ws_size,
                              hipStream_t stream) {
  (void)in_sizes; (void)n_in; (void)out_size; (void)ws_size;
  const float* dataA  = (const float*)d_in[0];
  const float* posA   = (const float*)d_in[1];
  const float* timeA  = (const float*)d_in[2];
  const float* varsA  = (const float*)d_in[3];
  const float* dtp    = (const float*)d_in[4];
  const int*   eidx   = (const int*)d_in[5];
  const int*   batch  = (const int*)d_in[6];
  const float* enc_w1 = (const float*)d_in[7];
  const float* enc_b1 = (const float*)d_in[8];
  const float* enc_w2 = (const float*)d_in[9];
  const float* enc_b2 = (const float*)d_in[10];
  const float* msg1_w = (const float*)d_in[11];
  const float* msg1_b = (const float*)d_in[12];
  const float* msg2_w = (const float*)d_in[13];
  const float* msg2_b = (const float*)d_in[14];
  const float* upd1_w = (const float*)d_in[15];
  const float* upd1_b = (const float*)d_in[16];
  const float* upd2_w = (const float*)d_in[17];
  const float* upd2_b = (const float*)d_in[18];
  const float* dec1_w = (const float*)d_in[19];
  const float* dec1_b = (const float*)d_in[20];
  const float* dec2_w = (const float*)d_in[21];
  const float* dec2_b = (const float*)d_in[22];
  float* outp = (float*)d_out;

  // allow 270KB dynamic LDS (WGP has 320KB on gfx1250)
  hipFuncSetAttribute((const void*)msg_kernel, hipFuncAttributeMaxDynamicSharedMemorySize, SH_BYTES);
  hipFuncSetAttribute((const void*)upd_kernel, hipFuncAttributeMaxDynamicSharedMemorySize, SH_BYTES);

  // ---- workspace layout (bf16 weights first) ----
  bf16_t* wb = (bf16_t*)d_ws;
  bf16_t* encW1 = wb + 0;                 // 176*32            = 5632
  bf16_t* encW2 = wb + 5632;              // 176*192           = 33792
  bf16_t* msg1W = wb + 39424;             // 6 * 176*384       = 405504
  bf16_t* msg2W = wb + 444928;            // 6 * 176*192       = 202752
  bf16_t* upd1W = wb + 647680;            // 6 * 176*384       = 405504
  bf16_t* upd2W = wb + 1053184;           // 6 * 176*192 ; end = 1255936 elems
  char* fb = (char*)d_ws + 2511872;       // 256B aligned
  bf16_t* hbf = (bf16_t*)fb;              // N * 176 bf16 = 35,200,000 B
  float* h    = (float*)(fb + 35200000);
  float* agg  = h   + (size_t)N_NODES * EDIM;
  float* hu   = agg + (size_t)N_NODES * EDIM;
  float* deg  = hu  + (size_t)N_NODES * EDIM;
  float* ssum = deg + N_NODES;
  float* ssq  = ssum + NGRP * EDIM;
  float* mean = ssq  + NGRP * EDIM;
  float* rstd = mean + NGRP * EDIM;
  float* cnt  = rstd + NGRP * EDIM;
  unsigned int* pmax = (unsigned int*)(cnt + NGRP);

  const int* esrc = eidx;
  const int* edst = eidx + N_EDGES;

  // ---- weight prep (bf16, transposed, padded; bias folded at bias_k) ----
  {
    int t;
    t = 1 * NPAD * KENC;    prep_kernel<<<(t + 255) / 256, 256, 0, stream>>>(enc_w1, enc_b1, encW1, 8,   KENC, 8,   t, 0);
    t = 1 * NPAD * KMID;    prep_kernel<<<(t + 255) / 256, 256, 0, stream>>>(enc_w2, enc_b2, encW2, 164, KMID, 176, t, 0);
    t = NLAY * NPAD * KBIG; prep_kernel<<<(t + 255) / 256, 256, 0, stream>>>(msg1_w, msg1_b, msg1W, 336, KBIG, 360, t, 1);
    t = NLAY * NPAD * KMID; prep_kernel<<<(t + 255) / 256, 256, 0, stream>>>(msg2_w, msg2_b, msg2W, 164, KMID, 176, t, 0);
    t = NLAY * NPAD * KBIG; prep_kernel<<<(t + 255) / 256, 256, 0, stream>>>(upd1_w, upd1_b, upd1W, 330, KBIG, 360, t, 1);
    t = NLAY * NPAD * KMID; prep_kernel<<<(t + 255) / 256, 256, 0, stream>>>(upd2_w, upd2_b, upd2W, 164, KMID, 176, t, 0);
  }

  // ---- degree, counts, pos-max ----
  zero_kernel<<<512, 256, 0, stream>>>(deg, N_NODES);
  zero_kernel<<<1, 64, 0, stream>>>(cnt, NGRP + 1);    // cnt + pmax (float 0 == uint 0)
  posmax_kernel<<<(N_NODES + 255) / 256, 256, 0, stream>>>(posA, pmax);
  deg_kernel<<<(N_EDGES + 255) / 256, 256, 0, stream>>>(edst, deg);
  cnt_kernel<<<(N_NODES + 255) / 256, 256, 0, stream>>>(batch, cnt);

  // ---- encoder ----
  enc_kernel<<<N_NODES / 32, 64, 0, stream>>>(dataA, posA, timeA, varsA,
                                              encW1, encW2, h, hbf, pmax);

  // ---- 6 message-passing layers ----
  for (int l = 0; l < NLAY; ++l) {
    zero_kernel<<<2048, 256, 0, stream>>>(agg, N_NODES * EDIM);
    zero_kernel<<<32, 256, 0, stream>>>(ssum, 2 * NGRP * EDIM);   // ssum + ssq contiguous
    msg_kernel<<<512, 128, SH_BYTES, stream>>>(
        hbf, dataA, posA, timeA, varsA, esrc, edst,
        msg1W + (size_t)l * NPAD * KBIG,
        msg2W + (size_t)l * NPAD * KMID, agg, pmax);
    upd_kernel<<<128, 128, SH_BYTES, stream>>>(
        h, hbf, agg, deg, timeA, varsA, batch,
        upd1W + (size_t)l * NPAD * KBIG,
        upd2W + (size_t)l * NPAD * KMID, hu, ssum, ssq);
    normstat_kernel<<<(NGRP * EDIM + 255) / 256, 256, 0, stream>>>(ssum, ssq, cnt, mean, rstd);
    applynorm_kernel<<<(N_NODES * HPAD + 255) / 256, 256, 0, stream>>>(hu, batch, mean, rstd, h, hbf);
  }

  // ---- decoder ----
  dec_kernel<<<(N_NODES + 255) / 256, 256, 0, stream>>>(h, dataA, dtp,
                                                        dec1_w, dec1_b, dec2_w, dec2_b, outp);
}
